// GAT_51599737094367
// MI455X (gfx1250) — matrix operand order repaired and checked by reading, not compiled, not run on hardware
//
#include <hip/hip_runtime.h>
#include <hip/hip_bf16.h>

typedef __attribute__((ext_vector_type(16))) _Float16 v16h;
typedef __attribute__((ext_vector_type(8)))  float    v8f;

#define NEG_SLOPE 0.2f

// ---------- monotone float<->uint map for atomic float max ----------
__device__ __forceinline__ unsigned fmax_map(float f) {
    unsigned u = __float_as_uint(f);
    return (u & 0x80000000u) ? ~u : (u | 0x80000000u);
}
__device__ __forceinline__ float fmax_unmap(unsigned u) {
    return (u & 0x80000000u) ? __uint_as_float(u & 0x7FFFFFFFu)
                             : __uint_as_float(~u);
}

// ---------- fragment-layout index helpers (ISA 7.12.2, 16-bit operands) ----------
// A (MxK=128): per m-tile, per kt (K=32 step), lane holds row (lane&15),
//   halves i=0..15 -> K = kt*32 + (lane>>4)*8 + (i&7) + (i>>3)*16
__device__ __forceinline__ size_t a_frag_idx(int row, int col) {
    int kt  = col >> 5;
    int rem = col & 31;
    int hi  = (rem >> 3) & 1;
    int i   = (rem & 7) + ((rem >> 4) << 3);
    int lane = hi * 16 + (row & 15);
    return ((size_t)((row >> 4) * 4 + kt) * 32 + lane) * 16 + i;
}
// B (128 x Ncols): per n-tile, per kt, lane holds col (lane&15),
//   halves i=0..15 -> K = kt*32 + (lane>>4)*8 + (i&7) + (i>>3)*16
__device__ __forceinline__ size_t b_frag_idx(int k, int n) {
    int nt = n >> 4;
    int kt = k >> 5;
    int krem = k & 31;
    int lane = ((krem >> 3) & 1) * 16 + (n & 15);
    return ((size_t)(nt * 4 + kt) * 32 + lane) * 16 + ((krem & 7) + ((krem >> 4) << 3));
}

// ---------- utility kernels ----------
__global__ void k_zero_u32(unsigned* __restrict__ p, int n) {
    int i = blockIdx.x * blockDim.x + threadIdx.x;
    if (i < n) p[i] = 0u;
}

// convert + pack A operand (M x 128 f32, row-major) into fragment layout
__global__ void k_pack_A(const float* __restrict__ X, _Float16* __restrict__ Ap, int total) {
    int i = blockIdx.x * blockDim.x + threadIdx.x;
    if (i >= total) return;
    int row = i >> 7, col = i & 127;
    Ap[a_frag_idx(row, col)] = (_Float16)X[i];
}

// convert + pack B operand (128 x Ncols f32, row-major) into fragment layout
__global__ void k_pack_B(const float* __restrict__ W, _Float16* __restrict__ Bp, int Ncols) {
    int j = blockIdx.x * blockDim.x + threadIdx.x;
    if (j >= 128 * Ncols) return;
    int k = j / Ncols, n = j - k * Ncols;
    Bp[b_frag_idx(k, n)] = (_Float16)W[j];
}

// ---------- WMMA GEMM: C[M x Ncols] = A[M x 128] * B[128 x Ncols] + bias ----------
// Operands pre-packed in fragment layout. One wave -> 16x32 output (2 n-tiles).
// blockDim = 256 (8 waves). M % 16 == 0, Ncols % 32 == 0.
__global__ void k_gemm_wmma(const v16h* __restrict__ Ap,
                            const v16h* __restrict__ Bp,
                            const float* __restrict__ bias,
                            float* __restrict__ C,
                            int M, int Ncols)
{
    const int lane   = threadIdx.x & 31;
    const int waveId = blockIdx.x * (blockDim.x >> 5) + (threadIdx.x >> 5);
    const int npair  = Ncols >> 5;
    const int mt     = waveId / npair;
    const int np     = waveId - mt * npair;
    if (mt >= (M >> 4)) return;

    const v16h* Abase = Ap + (size_t)mt * 4 * 32 + lane;
    const v16h* B0    = Bp + (size_t)(np * 2 + 0) * 4 * 32 + lane;
    const v16h* B1    = Bp + (size_t)(np * 2 + 1) * 4 * 32 + lane;

    v8f acc0 = {}, acc1 = {};
#pragma unroll
    for (int kt = 0; kt < 4; ++kt) {
        v16h a  = Abase[kt * 32];
        v16h b0 = B0[kt * 32];
        v16h b1 = B1[kt * 32];
        acc0 = __builtin_amdgcn_wmma_f32_16x16x32_f16(false, a, false, b0,
                                                      (short)0, acc0, false, false);
        acc1 = __builtin_amdgcn_wmma_f32_16x16x32_f16(false, a, false, b1,
                                                      (short)0, acc1, false, false);
    }

    const int hi = lane >> 4;
    const int c0 = np * 32 + (lane & 15);
    const int c1 = c0 + 16;
    const float bv0 = bias[c0], bv1 = bias[c1];
#pragma unroll
    for (int i = 0; i < 8; ++i) {          // C/D: row = i + 8*hi, col = lane&15
        int row = mt * 16 + hi * 8 + i;
        C[(size_t)row * Ncols + c0] = acc0[i] + bv0;
        C[(size_t)row * Ncols + c1] = acc1[i] + bv1;
    }
}

// ---------- layer 1 edge pass A: logits + atomic max (128 thr/edge) ----------
__global__ void k_edge_max_l1(const float* __restrict__ XS, const float* __restrict__ XD,
                              const float* __restrict__ att,
                              const long long* __restrict__ ei,
                              int E, int N, unsigned* __restrict__ mU,
                              float* __restrict__ elog)
{
    const int e = blockIdx.x * 2 + (threadIdx.x >> 7);
    const int t = threadIdx.x & 127;
    if (e >= E + N) return;
    __builtin_prefetch(&ei[e + 8192], 0, 0);
    int s, d;
    if (e < E) { s = (int)ei[e]; d = (int)ei[E + e]; } else { s = d = e - E; }

    float v = XS[(size_t)s * 128 + t] + XD[(size_t)d * 128 + t];
    v = v > 0.f ? v : NEG_SLOPE * v;
    float p = att[t] * v;
    p += __shfl_xor(p, 1); p += __shfl_xor(p, 2);
    p += __shfl_xor(p, 4); p += __shfl_xor(p, 8);     // per-head (16-lane) sum
    if ((t & 15) == 0) {
        const int h = t >> 4;
        elog[(size_t)e * 8 + h] = p;
        atomicMax(&mU[(size_t)d * 8 + h], fmax_map(p));
    }
}

// ---------- layer 1 edge pass B: ex from cached logit, scatter-add ----------
__global__ void k_edge_acc_l1(const float* __restrict__ XS,
                              const float* __restrict__ att,
                              const long long* __restrict__ ei,
                              int E, int N,
                              const unsigned* __restrict__ mU,
                              const float* __restrict__ elog,
                              float* __restrict__ denom, float* __restrict__ numer)
{
    const int e = blockIdx.x * 2 + (threadIdx.x >> 7);
    const int t = threadIdx.x & 127;
    if (e >= E + N) return;
    __builtin_prefetch(&ei[e + 8192], 0, 0);
    int s, d;
    if (e < E) { s = (int)ei[e]; d = (int)ei[E + e]; } else { s = d = e - E; }

    const int h = t >> 4;
    float lg = elog[(size_t)e * 8 + h];
    float mx = fmax_unmap(mU[(size_t)d * 8 + h]);
    float ex = __expf(lg - mx);
    float xsv = XS[(size_t)s * 128 + t];
    if ((t & 15) == 0) atomicAdd(&denom[(size_t)d * 8 + h], ex);
    atomicAdd(&numer[(size_t)d * 128 + t], ex * xsv);
}

// ---------- layer 1 finalize: numer/denom + bias, ELU, store packed f16 ----------
__global__ void k_node_fin_l1(const float* __restrict__ numer, const float* __restrict__ denom,
                              const float* __restrict__ bias,
                              _Float16* __restrict__ h1p, int N)
{
    int i = blockIdx.x * blockDim.x + threadIdx.x;
    if (i >= N * 128) return;
    int node = i >> 7, t = i & 127, h = t >> 4;
    float v = numer[i] / denom[(size_t)node * 8 + h] + bias[t];
    v = v > 0.f ? v : (__expf(v) - 1.0f);             // ELU
    h1p[a_frag_idx(node, t)] = (_Float16)v;
}

// ---------- layer 2 edge pass A: 1 wave/edge (heads=1, C=32) ----------
__global__ void k_edge_max_l2(const float* __restrict__ XS, const float* __restrict__ XD,
                              const float* __restrict__ att,
                              const long long* __restrict__ ei,
                              int E, int N, unsigned* __restrict__ mU,
                              float* __restrict__ elog)
{
    const int e = blockIdx.x * 8 + (threadIdx.x >> 5);
    const int c = threadIdx.x & 31;
    if (e >= E + N) return;
    __builtin_prefetch(&ei[e + 16384], 0, 0);
    int s, d;
    if (e < E) { s = (int)ei[e]; d = (int)ei[E + e]; } else { s = d = e - E; }

    float v = XS[(size_t)s * 32 + c] + XD[(size_t)d * 32 + c];
    v = v > 0.f ? v : NEG_SLOPE * v;
    float p = att[c] * v;
    p += __shfl_xor(p, 1); p += __shfl_xor(p, 2); p += __shfl_xor(p, 4);
    p += __shfl_xor(p, 8); p += __shfl_xor(p, 16);
    if (c == 0) {
        elog[e] = p;
        atomicMax(&mU[d], fmax_map(p));
    }
}

__global__ void k_edge_acc_l2(const float* __restrict__ XS,
                              const long long* __restrict__ ei,
                              int E, int N,
                              const unsigned* __restrict__ mU,
                              const float* __restrict__ elog,
                              float* __restrict__ denom, float* __restrict__ numer)
{
    const int e = blockIdx.x * 8 + (threadIdx.x >> 5);
    const int c = threadIdx.x & 31;
    if (e >= E + N) return;
    __builtin_prefetch(&ei[e + 16384], 0, 0);
    int s, d;
    if (e < E) { s = (int)ei[e]; d = (int)ei[E + e]; } else { s = d = e - E; }

    float ex = __expf(elog[e] - fmax_unmap(mU[d]));
    float xsv = XS[(size_t)s * 32 + c];
    if (c == 0) atomicAdd(&denom[d], ex);
    atomicAdd(&numer[(size_t)d * 32 + c], ex * xsv);
}

// ---------- layer 2 finalize + log_softmax: 1 wave / node ----------
__global__ void k_node_out_l2(const float* __restrict__ numer, const float* __restrict__ denom,
                              const float* __restrict__ bias,
                              float* __restrict__ out, int N)
{
    const int node = blockIdx.x * 8 + (threadIdx.x >> 5);
    const int c    = threadIdx.x & 31;
    if (node >= N) return;
    float v = numer[(size_t)node * 32 + c] / denom[node] + bias[c];
    float mx = v;
    mx = fmaxf(mx, __shfl_xor(mx, 1));  mx = fmaxf(mx, __shfl_xor(mx, 2));
    mx = fmaxf(mx, __shfl_xor(mx, 4));  mx = fmaxf(mx, __shfl_xor(mx, 8));
    mx = fmaxf(mx, __shfl_xor(mx, 16));
    float s = __expf(v - mx);
    s += __shfl_xor(s, 1); s += __shfl_xor(s, 2); s += __shfl_xor(s, 4);
    s += __shfl_xor(s, 8); s += __shfl_xor(s, 16);
    out[(size_t)node * 32 + c] = v - mx - __logf(s);
}

// ---------------------------------------------------------------------------
extern "C" void kernel_launch(void* const* d_in, const int* in_sizes, int n_in,
                              void* d_out, int out_size, void* d_ws, size_t ws_size,
                              hipStream_t stream)
{
    const float*     x    = (const float*)d_in[0];
    const long long* ei   = (const long long*)d_in[1];
    const float*     W1s  = (const float*)d_in[2];
    const float*     W1d  = (const float*)d_in[3];
    const float*     b1s  = (const float*)d_in[4];
    const float*     b1d  = (const float*)d_in[5];
    const float*     att1 = (const float*)d_in[6];
    const float*     bia1 = (const float*)d_in[7];
    const float*     W2s  = (const float*)d_in[8];
    const float*     W2d  = (const float*)d_in[9];
    const float*     b2s  = (const float*)d_in[10];
    const float*     b2d  = (const float*)d_in[11];
    const float*     att2 = (const float*)d_in[12];
    const float*     bia2 = (const float*)d_in[13];
    float* out = (float*)d_out;

    const int N = in_sizes[0] / 128;
    const int E = in_sizes[1] / 2;
    const int Etot = E + N;

    // ---- workspace carve-out (256B aligned) ----
    char* ws = (char*)d_ws;
    size_t off = 0;
    auto carve = [&](size_t bytes) -> char* {
        char* p = ws + off;
        off += (bytes + 255) & ~(size_t)255;
        return p;
    };
    _Float16* Xp   = (_Float16*)carve((size_t)N * 128 * 2);   // packed A, layer 1
    _Float16* W1sp = (_Float16*)carve(128 * 128 * 2);         // packed B
    _Float16* W1dp = (_Float16*)carve(128 * 128 * 2);
    _Float16* W2sp = (_Float16*)carve(128 * 32 * 2);
    _Float16* W2dp = (_Float16*)carve(128 * 32 * 2);
    float*    XS1  = (float*)carve((size_t)N * 128 * 4);
    float*    XD1  = (float*)carve((size_t)N * 128 * 4);
    unsigned* m1   = (unsigned*)carve((size_t)N * 8 * 4);
    float*    den1 = (float*)carve((size_t)N * 8 * 4);
    float*    num1 = (float*)carve((size_t)N * 128 * 4);
    _Float16* H1p  = (_Float16*)carve((size_t)N * 128 * 2);   // packed A, layer 2
    float*    elog1= (float*)carve((size_t)Etot * 8 * 4);     // cached edge logits
    // layer-2 buffers alias dead layer-1 buffers:
    float*    XS2   = XS1;              // N*32 f32
    float*    XD2   = XD1;              // N*32 f32
    unsigned* m2    = m1;               // N u32
    float*    den2  = den1;             // N f32
    float*    num2  = num1;             // N*32 f32
    float*    elog2 = elog1;            // Etot f32

    const int T = 256;
    // ---- convert + pack operands ----
    k_pack_A<<<(N * 128 + T - 1) / T, T, 0, stream>>>(x, Xp, N * 128);
    k_pack_B<<<(128 * 128 + T - 1) / T, T, 0, stream>>>(W1s, W1sp, 128);
    k_pack_B<<<(128 * 128 + T - 1) / T, T, 0, stream>>>(W1d, W1dp, 128);
    k_pack_B<<<(128 * 32  + T - 1) / T, T, 0, stream>>>(W2s, W2sp, 32);
    k_pack_B<<<(128 * 32  + T - 1) / T, T, 0, stream>>>(W2d, W2dp, 32);

    // ---- layer 1 GEMMs (WMMA) ----
    {
        int waves = (N / 16) * (128 / 32);
        int blocks = (waves + 7) / 8;
        k_gemm_wmma<<<blocks, T, 0, stream>>>((const v16h*)Xp, (const v16h*)W1sp, b1s, XS1, N, 128);
        k_gemm_wmma<<<blocks, T, 0, stream>>>((const v16h*)Xp, (const v16h*)W1dp, b1d, XD1, N, 128);
    }

    // ---- zero layer-1 accumulators ----
    k_zero_u32<<<(N * 8   + T - 1) / T, T, 0, stream>>>(m1, N * 8);
    k_zero_u32<<<(N * 8   + T - 1) / T, T, 0, stream>>>((unsigned*)den1, N * 8);
    k_zero_u32<<<(N * 128 + T - 1) / T, T, 0, stream>>>((unsigned*)num1, N * 128);

    // ---- layer 1 edge passes ----
    {
        int blocks = (Etot + 1) / 2;
        k_edge_max_l1<<<blocks, T, 0, stream>>>(XS1, XD1, att1, ei, E, N, m1, elog1);
        k_edge_acc_l1<<<blocks, T, 0, stream>>>(XS1, att1, ei, E, N, m1, elog1, den1, num1);
    }
    k_node_fin_l1<<<(N * 128 + T - 1) / T, T, 0, stream>>>(num1, den1, bia1, H1p, N);

    // ---- layer 2 GEMMs (WMMA) ----
    {
        int waves = (N / 16) * (32 / 32);
        int blocks = (waves + 7) / 8;
        k_gemm_wmma<<<blocks, T, 0, stream>>>((const v16h*)H1p, (const v16h*)W2sp, b2s, XS2, N, 32);
        k_gemm_wmma<<<blocks, T, 0, stream>>>((const v16h*)H1p, (const v16h*)W2dp, b2d, XD2, N, 32);
    }

    // ---- zero layer-2 accumulators (aliased buffers) ----
    k_zero_u32<<<(N      + T - 1) / T, T, 0, stream>>>(m2, N);
    k_zero_u32<<<(N      + T - 1) / T, T, 0, stream>>>((unsigned*)den2, N);
    k_zero_u32<<<(N * 32 + T - 1) / T, T, 0, stream>>>((unsigned*)num2, N * 32);

    // ---- layer 2 edge passes ----
    {
        int blocks = (Etot + 7) / 8;
        k_edge_max_l2<<<blocks, T, 0, stream>>>(XS2, XD2, att2, ei, E, N, m2, elog2);
        k_edge_acc_l2<<<blocks, T, 0, stream>>>(XS2, ei, E, N, m2, elog2, den2, num2);
    }

    // ---- finalize + log_softmax ----
    k_node_out_l2<<<(N + 7) / 8, T, 0, stream>>>(num2, den2, bia2, out, N);
    (void)n_in; (void)out_size; (void)ws_size;
}
